// SelfAttention_61701500174831
// MI455X (gfx1250) — matrix operand order repaired and checked by reading, not compiled, not run on hardware
//
#include <hip/hip_runtime.h>
#include <hip/hip_bf16.h>

// ---------------------------------------------------------------------------
// CDNA5 (gfx1250) self-attention: bf16 WMMA + async global->LDS pipelining.
//   B=2, L=2048, D=2048, H=16, DH=128
// GEMMs: 128x128 block tile, 8 waves (4x2), 2x4 WMMA register blocking,
// double-buffered LDS fed by GLOBAL_LOAD_ASYNC_TO_LDS_B128 (ASYNCcnt).
// ---------------------------------------------------------------------------

#define Bsz 2
#define Lsz 2048
#define Dsz 2048
#define Hsz 16
#define DHsz 128

#define USE_ASYNC_COPY 1   // global_load_async_to_lds_b128 (+ s_wait_asynccnt)

typedef __bf16 bf16_t;
typedef __attribute__((ext_vector_type(16))) __bf16 v16bf;
typedef __attribute__((ext_vector_type(2)))  __bf16 bf16x2;
typedef __attribute__((ext_vector_type(8)))  float  v8f;

// --------------------------- async copy helpers ----------------------------
// GLOBAL_LOAD_ASYNC_TO_LDS_B128: per-lane 16B global->LDS, tracked by ASYNCcnt.
// LDS destination operand is a VGPR holding the LDS byte offset; for a generic
// pointer into __shared__, that is addr[31:0] (flat aperture rule).

__device__ __forceinline__ void cp_async_b128(const bf16_t* gsrc, bf16_t* ldst) {
#if USE_ASYNC_COPY
  unsigned ldsoff = (unsigned)(unsigned long long)ldst;
  asm volatile("global_load_async_to_lds_b128 %0, %1, off"
               :: "v"(ldsoff), "v"(gsrc)
               : "memory");
#else
  *(uint4*)ldst = *(const uint4*)gsrc;
#endif
}

// Wait until at most N async ops outstanding (in-order completion).
template <int N>
__device__ __forceinline__ void wait_async() {
#if USE_ASYNC_COPY
  asm volatile("s_wait_asynccnt %0" :: "i"(N) : "memory");
#endif
}

// --------------------------- WMMA helpers ----------------------------------

__device__ __forceinline__ v8f wmma_bf16(v16bf a, v16bf b, v8f c) {
  return __builtin_amdgcn_wmma_f32_16x16x32_bf16(
      /*neg_a=*/false, a, /*neg_b=*/false, b,
      /*c_mod=*/(short)0, c, /*reuse_a=*/false, /*reuse_b=*/false);
}

// A fragment: A row-major [16, K], element (m,k) = A[m*lda + k]
__device__ __forceinline__ v16bf load_frag_a(const bf16_t* A, int lda) {
  const int lane = threadIdx.x & 31;
  const int m = lane & 15;
  const int kb = (lane & 16) ? 16 : 0;
  const bf16_t* row = A + m * lda + kb;
  v16bf f;
#pragma unroll
  for (int v = 0; v < 8; ++v) {
    const int k = 2 * v;
    bf16x2 p = *(const bf16x2*)(row + k);
    f[2 * v] = p.x;
    f[2 * v + 1] = p.y;
  }
  return f;
}

// B fragment from W stored [N, K]: element (k,n) = W[n*ldw + k]
__device__ __forceinline__ v16bf load_frag_b_nk(const bf16_t* W, int ldw) {
  const int lane = threadIdx.x & 31;
  const int n = lane & 15;
  const int kb = (lane & 16) ? 16 : 0;
  const bf16_t* col = W + n * ldw + kb;
  v16bf f;
#pragma unroll
  for (int v = 0; v < 8; ++v) {
    bf16x2 p = *(const bf16x2*)(col + 2 * v);
    f[2 * v] = p.x;
    f[2 * v + 1] = p.y;
  }
  return f;
}

// B fragment from Bm stored [K, N]: element (k,n) = Bm[k*ldb + n]
__device__ __forceinline__ v16bf load_frag_b_kn(const bf16_t* Bm, int ldb) {
  const int lane = threadIdx.x & 31;
  const int n = lane & 15;
  const int kb = (lane & 16) ? 16 : 0;
  v16bf f;
#pragma unroll
  for (int v = 0; v < 16; ++v) f[v] = Bm[(size_t)(kb + v) * ldb + n];
  return f;
}

// --------------------------- convert f32 -> bf16 ---------------------------

__global__ __launch_bounds__(256) void sa_cvt_bf16(const float* __restrict__ src,
                                                   bf16_t* __restrict__ dst,
                                                   size_t n) {
  size_t i = (size_t)blockIdx.x * blockDim.x + threadIdx.x;
  if (i < n) dst[i] = (bf16_t)src[i];
}

// --------------------------- GEMM core -------------------------------------
// Block tile 128x128, 256 threads (8 waves as 4 row-groups x 2 col-groups).
// Wave computes 32x64 (2 A-frags x 4 B-frags = 8 WMMA per K-step of 32).
// Double-buffered LDS, async prefetch (4 async ops per wave per stage).

__device__ __forceinline__ void gemm_stage(const bf16_t* __restrict__ A,
                                           const bf16_t* __restrict__ Bw,
                                           bf16_t* lA, bf16_t* lB,
                                           int m0, int n0, int k0, int tid) {
  const int row = tid >> 1, half = tid & 1;
  const bf16_t* asrc = &A[(size_t)(m0 + row) * Dsz + k0 + half * 16];
  bf16_t* adst = &lA[row * 32 + half * 16];
  cp_async_b128(asrc, adst);
  cp_async_b128(asrc + 8, adst + 8);
  const bf16_t* bsrc = &Bw[(size_t)(n0 + row) * Dsz + k0 + half * 16];
  bf16_t* bdst = &lB[row * 32 + half * 16];
  cp_async_b128(bsrc, bdst);
  cp_async_b128(bsrc + 8, bdst + 8);
}

__device__ __forceinline__ void gemm_tile(const bf16_t* __restrict__ A,
                                          const bf16_t* __restrict__ Bw,
                                          bf16_t (*lA)[128 * 32],
                                          bf16_t (*lB)[128 * 32],
                                          int m0, int n0, int tid,
                                          int wr, int wc, v8f acc[2][4]) {
  gemm_stage(A, Bw, lA[0], lB[0], m0, n0, 0, tid);
  for (int k0 = 0; k0 < Dsz; k0 += 32) {
    const int cur = (k0 >> 5) & 1;
    if (k0 + 32 < Dsz) {
      gemm_stage(A, Bw, lA[cur ^ 1], lB[cur ^ 1], m0, n0, k0 + 32, tid);
      wait_async<4>();   // oldest 4 (current tile) landed; prefetch in flight
    } else {
      wait_async<0>();
    }
    __syncthreads();
    // hoist all 6 fragment loads, then 8 back-to-back WMMAs
    v16bf a0 = load_frag_a(&lA[cur][(wr * 32) * 32], 32);
    v16bf a1 = load_frag_a(&lA[cur][(wr * 32 + 16) * 32], 32);
    v16bf b0 = load_frag_b_nk(&lB[cur][(wc * 64) * 32], 32);
    v16bf b1 = load_frag_b_nk(&lB[cur][(wc * 64 + 16) * 32], 32);
    v16bf b2 = load_frag_b_nk(&lB[cur][(wc * 64 + 32) * 32], 32);
    v16bf b3 = load_frag_b_nk(&lB[cur][(wc * 64 + 48) * 32], 32);
    acc[0][0] = wmma_bf16(a0, b0, acc[0][0]);
    acc[1][0] = wmma_bf16(a1, b0, acc[1][0]);
    acc[0][1] = wmma_bf16(a0, b1, acc[0][1]);
    acc[1][1] = wmma_bf16(a1, b1, acc[1][1]);
    acc[0][2] = wmma_bf16(a0, b2, acc[0][2]);
    acc[1][2] = wmma_bf16(a1, b2, acc[1][2]);
    acc[0][3] = wmma_bf16(a0, b3, acc[0][3]);
    acc[1][3] = wmma_bf16(a1, b3, acc[1][3]);
    __syncthreads();   // all ds reads done before buffer is re-staged
  }
}

// --------------------------- QKV projection --------------------------------

__global__ __launch_bounds__(256) void sa_gemm_qkv(const bf16_t* __restrict__ xb,
                                                   const bf16_t* __restrict__ wq,
                                                   bf16_t* __restrict__ qkv) {
  __shared__ __align__(16) bf16_t lA[2][128 * 32];
  __shared__ __align__(16) bf16_t lB[2][128 * 32];
  const int m0 = blockIdx.x * 128;
  const int n0 = blockIdx.y * 128;
  const int tid = threadIdx.x;
  const int wr = (tid >> 5) & 3;
  const int wc = tid >> 7;
  v8f acc[2][4] = {};
  gemm_tile(xb, wq, lA, lB, m0, n0, tid, wr, wc, acc);

  // scatter-store: n in [0,6144) -> {q,k,v}[b,h,l,dh], bf16
  const int lane = tid & 31;
  const int n_l = lane & 15;
  const int mb = (lane & 16) ? 8 : 0;
#pragma unroll
  for (int rt = 0; rt < 2; ++rt) {
#pragma unroll
    for (int c = 0; c < 4; ++c) {
      const int n_g = n0 + wc * 64 + c * 16 + n_l;
      const int sec = n_g >> 11;      // 0=q 1=k 2=v
      const int within = n_g & 2047;
      const int h = within >> 7;
      const int dh = within & 127;
#pragma unroll
      for (int r = 0; r < 8; ++r) {
        const int m_g = m0 + wr * 32 + rt * 16 + mb + r;
        const int b = m_g >> 11;      // / L
        const int l = m_g & 2047;
        qkv[(size_t)sec * ((size_t)Bsz * Hsz * Lsz * DHsz) +
            ((size_t)(b * Hsz + h) * Lsz + l) * DHsz + dh] = (bf16_t)acc[rt][c][r];
      }
    }
  }
}

// --------------------------- output projection -----------------------------

__global__ __launch_bounds__(256) void sa_gemm_out(const bf16_t* __restrict__ yb,
                                                   const bf16_t* __restrict__ wo,
                                                   const float* __restrict__ bias,
                                                   float* __restrict__ out) {
  __shared__ __align__(16) bf16_t lA[2][128 * 32];
  __shared__ __align__(16) bf16_t lB[2][128 * 32];
  const int m0 = blockIdx.x * 128;
  const int n0 = blockIdx.y * 128;
  const int tid = threadIdx.x;
  const int wr = (tid >> 5) & 3;
  const int wc = tid >> 7;
  v8f acc[2][4] = {};
  gemm_tile(yb, wo, lA, lB, m0, n0, tid, wr, wc, acc);

  const int lane = tid & 31;
  const int n_l = lane & 15;
  const int mb = (lane & 16) ? 8 : 0;
#pragma unroll
  for (int rt = 0; rt < 2; ++rt) {
#pragma unroll
    for (int c = 0; c < 4; ++c) {
      const int n_g = n0 + wc * 64 + c * 16 + n_l;
      const float bv = bias[n_g];
#pragma unroll
      for (int r = 0; r < 8; ++r) {
        const int m_g = m0 + wr * 32 + rt * 16 + mb + r;
        out[(size_t)m_g * Dsz + n_g] = acc[rt][c][r] + bv;
      }
    }
  }
}

// --------------------------- RoPE (in place on q,k) ------------------------

__global__ __launch_bounds__(256) void sa_rope(bf16_t* __restrict__ q,
                                               bf16_t* __restrict__ k) {
  const size_t per = (size_t)Bsz * Hsz * Lsz * 64;
  const size_t i = (size_t)blockIdx.x * blockDim.x + threadIdx.x;
  bf16_t* ptr = (i < per) ? q : k;
  const size_t j = (i < per) ? i : (i - per);
  const int f = (int)(j & 63);
  const size_t bhl = j >> 6;
  const int l = (int)(bhl & (Lsz - 1));
  const size_t base = bhl * DHsz;
  const float inv = __powf(10000.0f, -((float)(2 * f)) * (1.0f / 128.0f));
  float s, c;
  __sincosf((float)l * inv, &s, &c);
  const float x1 = (float)ptr[base + f];
  const float x2 = (float)ptr[base + f + 64];
  ptr[base + f]      = (bf16_t)(x1 * c - x2 * s);
  ptr[base + f + 64] = (bf16_t)(x2 * c + x1 * s);
}

// --------------------------- Flash attention -------------------------------
// grid (B*H, L/64), 128 threads (4 waves). Wave w owns query rows
// [q0+16w, q0+16w+16) x DH=128. K/V tiles async-staged into LDS each key
// block; P re-laid-out via per-wave LDS region.

__global__ __launch_bounds__(128) void sa_flash(const bf16_t* __restrict__ Q,
                                                const bf16_t* __restrict__ K,
                                                const bf16_t* __restrict__ V,
                                                bf16_t* __restrict__ Y) {
  __shared__ __align__(16) bf16_t lK[64 * DHsz];   // 16 KB
  __shared__ __align__(16) bf16_t lV[64 * DHsz];   // 16 KB
  __shared__ __align__(16) bf16_t lP[4][16 * 64];  //  8 KB
  const int bh = blockIdx.x;
  const int qb = blockIdx.y;
  const int b = bh >> 4;  // / H
  const int h = bh & 15;
  const int tid = threadIdx.x;
  const int w = tid >> 5;
  const int lane = tid & 31;
  const int n_l = lane & 15;
  const int mb = (lane & 16) ? 8 : 0;
  const int q0 = qb * 64;
  const int row0 = q0 + w * 16;
  const size_t headBase = (size_t)bh * Lsz * DHsz;
  const bf16_t* Qh = Q + headBase;
  const bf16_t* Kh = K + headBase;
  const bf16_t* Vh = V + headBase;

  v16bf qf[4];
#pragma unroll
  for (int kk = 0; kk < 4; ++kk)
    qf[kk] = load_frag_a(Qh + (size_t)row0 * DHsz + 32 * kk, DHsz);

  v8f acc[8] = {};
  float row_m[8], row_l[8];
#pragma unroll
  for (int r = 0; r < 8; ++r) { row_m[r] = -1e30f; row_l[r] = 0.0f; }

  const float scale = 0.08838834764831845f;  // 1/sqrt(128)

  for (int kb = 0; kb <= qb; ++kb) {
    const int key0 = kb * 64;
    // ---- async-stage K and V tiles (64x128 bf16 each, fully contiguous)
#pragma unroll
    for (int j = 0; j < 8; ++j) {
      const int off = j * 1024 + tid * 8;
      cp_async_b128(Kh + (size_t)key0 * DHsz + off, &lK[off]);
      cp_async_b128(Vh + (size_t)key0 * DHsz + off, &lV[off]);
    }
    wait_async<0>();
    __syncthreads();

    // ---- S(16x64) = Q(16x128) * K^T : hoist 4 fragments per score tile
    v8f s[4] = {};
#pragma unroll
    for (int t = 0; t < 4; ++t) {
      v16bf kf0 = load_frag_b_nk(&lK[(16 * t) * DHsz + 0], DHsz);
      v16bf kf1 = load_frag_b_nk(&lK[(16 * t) * DHsz + 32], DHsz);
      v16bf kf2 = load_frag_b_nk(&lK[(16 * t) * DHsz + 64], DHsz);
      v16bf kf3 = load_frag_b_nk(&lK[(16 * t) * DHsz + 96], DHsz);
      s[t] = wmma_bf16(qf[0], kf0, s[t]);
      s[t] = wmma_bf16(qf[1], kf1, s[t]);
      s[t] = wmma_bf16(qf[2], kf2, s[t]);
      s[t] = wmma_bf16(qf[3], kf3, s[t]);
    }
    // ---- scale + causal mask (branchless, uniform EXEC around WMMA)
#pragma unroll
    for (int t = 0; t < 4; ++t) {
      const int col = key0 + 16 * t + n_l;
#pragma unroll
      for (int r = 0; r < 8; ++r) {
        float v = s[t][r] * scale;
        if (col > row0 + mb + r) v = -1e30f;
        s[t][r] = v;
      }
    }
    // ---- online softmax: row stats replicated across each 16-lane half
#pragma unroll
    for (int r = 0; r < 8; ++r) {
      float pm = fmaxf(fmaxf(s[0][r], s[1][r]), fmaxf(s[2][r], s[3][r]));
#pragma unroll
      for (int off = 1; off < 16; off <<= 1)
        pm = fmaxf(pm, __shfl_xor(pm, off, 32));
      const float nm = fmaxf(row_m[r], pm);
      const float rescale = __expf(row_m[r] - nm);
      row_m[r] = nm;
      row_l[r] *= rescale;
#pragma unroll
      for (int c = 0; c < 8; ++c) acc[c][r] *= rescale;
    }
    // ---- P = exp(S-m); row sums; stage P in per-wave LDS (C-frag -> A-frag)
#pragma unroll
    for (int r = 0; r < 8; ++r) {
      float ps = 0.0f;
#pragma unroll
      for (int t = 0; t < 4; ++t) {
        const float e = __expf(s[t][r] - row_m[r]);
        ps += e;
        lP[w][(mb + r) * 64 + 16 * t + n_l] = (bf16_t)e;
      }
#pragma unroll
      for (int off = 1; off < 16; off <<= 1)
        ps += __shfl_xor(ps, off, 32);
      row_l[r] += ps;
    }
    // ---- O += P(16x64) * V(64x128), fragments batched ahead of WMMA chains
#pragma unroll
    for (int kk = 0; kk < 2; ++kk) {
      v16bf pf = load_frag_a(&lP[w][32 * kk], 64);
      v16bf vf0 = load_frag_b_kn(&lV[(32 * kk) * DHsz + 0],  DHsz);
      v16bf vf1 = load_frag_b_kn(&lV[(32 * kk) * DHsz + 16], DHsz);
      v16bf vf2 = load_frag_b_kn(&lV[(32 * kk) * DHsz + 32], DHsz);
      v16bf vf3 = load_frag_b_kn(&lV[(32 * kk) * DHsz + 48], DHsz);
      acc[0] = wmma_bf16(pf, vf0, acc[0]);
      acc[1] = wmma_bf16(pf, vf1, acc[1]);
      acc[2] = wmma_bf16(pf, vf2, acc[2]);
      acc[3] = wmma_bf16(pf, vf3, acc[3]);
      vf0 = load_frag_b_kn(&lV[(32 * kk) * DHsz + 64],  DHsz);
      vf1 = load_frag_b_kn(&lV[(32 * kk) * DHsz + 80],  DHsz);
      vf2 = load_frag_b_kn(&lV[(32 * kk) * DHsz + 96],  DHsz);
      vf3 = load_frag_b_kn(&lV[(32 * kk) * DHsz + 112], DHsz);
      acc[4] = wmma_bf16(pf, vf0, acc[4]);
      acc[5] = wmma_bf16(pf, vf1, acc[5]);
      acc[6] = wmma_bf16(pf, vf2, acc[6]);
      acc[7] = wmma_bf16(pf, vf3, acc[7]);
    }
    __syncthreads();   // lK/lV fully consumed before next block restages
  }

  // ---- normalize and store y [B, L, D] bf16 (heads merged)
  bf16_t* Yb = Y + (size_t)b * Lsz * Dsz + (size_t)h * DHsz;
#pragma unroll
  for (int c = 0; c < 8; ++c) {
#pragma unroll
    for (int r = 0; r < 8; ++r) {
      const float v = acc[c][r] / row_l[r];
      Yb[(size_t)(row0 + mb + r) * Dsz + 16 * c + n_l] = (bf16_t)v;
    }
  }
}

// --------------------------- host launcher ---------------------------------

extern "C" void kernel_launch(void* const* d_in, const int* in_sizes, int n_in,
                              void* d_out, int out_size, void* d_ws, size_t ws_size,
                              hipStream_t stream) {
  const float* x     = (const float*)d_in[0];  // [B, L, D]
  const float* w_qkv = (const float*)d_in[1];  // [3D, D]
  const float* w_out = (const float*)d_in[2];  // [D, D]
  const float* b_out = (const float*)d_in[3];  // [D]
  float* out = (float*)d_out;                  // [B, L, D] fp32

  const size_t nX    = (size_t)Bsz * Lsz * Dsz;
  const size_t nWqk  = (size_t)3 * Dsz * Dsz;
  const size_t nWo   = (size_t)Dsz * Dsz;
  const size_t nHead = (size_t)Bsz * Hsz * Lsz * DHsz;

  char* ws = (char*)d_ws;
  bf16_t* xb    = (bf16_t*)ws;  ws += nX * sizeof(bf16_t);
  bf16_t* wqkvb = (bf16_t*)ws;  ws += nWqk * sizeof(bf16_t);
  bf16_t* woutb = (bf16_t*)ws;  ws += nWo * sizeof(bf16_t);
  bf16_t* qkvb  = (bf16_t*)ws;  ws += 3 * nHead * sizeof(bf16_t);
  bf16_t* yb    = (bf16_t*)ws;  ws += nX * sizeof(bf16_t);
  bf16_t* qh = qkvb;
  bf16_t* kh = qkvb + nHead;
  bf16_t* vh = qkvb + 2 * nHead;

  sa_cvt_bf16<<<(unsigned)((nX + 255) / 256), 256, 0, stream>>>(x, xb, nX);
  sa_cvt_bf16<<<(unsigned)((nWqk + 255) / 256), 256, 0, stream>>>(w_qkv, wqkvb, nWqk);
  sa_cvt_bf16<<<(unsigned)((nWo + 255) / 256), 256, 0, stream>>>(w_out, woutb, nWo);

  sa_gemm_qkv<<<dim3(Bsz * Lsz / 128, 3 * Dsz / 128), 256, 0, stream>>>(xb, wqkvb, qkvb);

  {
    const size_t total = 2 * (size_t)Bsz * Hsz * Lsz * 64;
    sa_rope<<<(unsigned)(total / 256), 256, 0, stream>>>(qh, kh);
  }

  sa_flash<<<dim3(Bsz * Hsz, Lsz / 64), 128, 0, stream>>>(qh, kh, vh, yb);

  sa_gemm_out<<<dim3(Bsz * Lsz / 128, Dsz / 128), 256, 0, stream>>>(yb, woutb, b_out, out);
}
